// TransformerModel_40553081208955
// MI455X (gfx1250) — matrix operand order repaired and checked by reading, not compiled, not run on hardware
//
#include <hip/hip_runtime.h>
#include <hip/hip_bf16.h>
#include <math.h>

// ---------------------------------------------------------------------------
// Transformer forward for MI455X (gfx1250, wave32, WMMA).
// GEMMs: v_wmma_f32_16x16x32_f16 with f32 accumulate; A (and B for the K^T
// case) tiles are DMA'd into LDS with global_load_async_to_lds_b128 and
// double-buffered (ASYNCcnt pipeline); f32 -> f16 conversion happens during
// fragment build (v_cvt_pk_f16_f32 co-executes with WMMA).
// ---------------------------------------------------------------------------

#define DMODEL 512
#define NHEAD  8
#define HDIM   64          // DMODEL / NHEAD
#define NLAYER 6
#define DFFN   2048
#define VOCAB  32000
#define BATCH  4
#define SEQ    512
#define ROWS   (BATCH*SEQ) // 2048

typedef _Float16 v16h __attribute__((ext_vector_type(16)));
typedef _Float16 v8h  __attribute__((ext_vector_type(8)));
typedef _Float16 v4h  __attribute__((ext_vector_type(4)));
typedef float    v8f  __attribute__((ext_vector_type(8)));
typedef float    v4f  __attribute__((ext_vector_type(4)));

// ---- CDNA5 async global->LDS copy (ASYNCcnt) ------------------------------
#if defined(__has_builtin)
# if __has_builtin(__builtin_amdgcn_global_load_async_to_lds_b128)
#  define HAVE_ASYNC_LDS_BUILTIN 1
# endif
# if __has_builtin(__builtin_amdgcn_s_wait_asynccnt)
#  define HAVE_WAIT_ASYNC_BUILTIN 1
# endif
#endif

// Builtin prototype (from hipcc diagnostic): param0 is
//   '__attribute__((__vector_size__(4*sizeof(int)))) int __device__ *'
// i.e. a v4i32 pointer in the global address space; LDS side analogous (AS3).
typedef int vint4 __attribute__((vector_size(16)));
typedef __attribute__((address_space(1))) vint4 as1_vint4;
typedef __attribute__((address_space(3))) vint4 as3_vint4;

__device__ __forceinline__ void async_b128_to_lds(const float* g, void* lds)
{
    // Cast via integers: low 32 bits of a generic LDS pointer are the LDS byte
    // offset (ISA 10.2 aperture rules); AS1 address equals the generic one.
#ifdef HAVE_ASYNC_LDS_BUILTIN
    __builtin_amdgcn_global_load_async_to_lds_b128(
        (as1_vint4*)(unsigned long long)(const void*)g,
        (as3_vint4*)(unsigned)(unsigned long long)lds,
        0, 0);
#else
    asm volatile("global_load_async_to_lds_b128 %0, %1, off"
                 :: "v"((unsigned)(unsigned long long)lds), "v"(g)
                 : "memory");
#endif
}

__device__ __forceinline__ void wait_asynccnt0()
{
#ifdef HAVE_WAIT_ASYNC_BUILTIN
    __builtin_amdgcn_s_wait_asynccnt(0);
#else
    asm volatile("s_wait_asynccnt 0x0" ::: "memory");
#endif
}

// ---------------------------------------------------------------------------
// WMMA GEMM: C[M,N] = A[M,K] * B  (+bias) (+relu)
//   BT=false : B is [K,N] row-major (transposed into LDS as f16 [N][K])
//   BT=true  : B is [N,K] row-major (A * B^T; async f32 copy like A)
// Block tile 64x64, BK=32, 128 threads = 4 waves in 2x2, each wave 2x2 WMMAs.
// Double-buffered LDS; prefetch of step t+1 overlaps compute of step t.
// Requires M%64==0, N%64==0, K%32==0 (true for every GEMM in this model).
// ---------------------------------------------------------------------------
template <bool BT, int EPI>   // EPI: 0=none, 1=+bias, 2=+bias+relu
__global__ __launch_bounds__(128)
void gemm_wmma_kernel(const float* __restrict__ A, const float* __restrict__ B,
                      const float* __restrict__ bias, float* __restrict__ C,
                      int K, int lda, int ldb, int ldc, int nheads,
                      long sAb, long sAh, long sBb, long sBh, long sCb, long sCh)
{
    constexpr int BM = 64, BN = 64, BK = 32;
    constexpr int AST  = 36;            // f32 tile row stride (floats, 144B: 16B-aligned rows)
    constexpr int ABUF = BM * AST;      // floats per A buffer
    constexpr int BSTH = 40;            // f16 transposed tile row stride (halfs)

    __shared__ float As[2 * ABUF];
    __shared__ __align__(16) unsigned char Braw[BT ? (2 * BN * AST * 4)
                                                   : (2 * BN * BSTH * 2)];
    float*    Bf = (float*)Braw;        // BT path (f32, [n][k])
    _Float16* Bh = (_Float16*)Braw;     // !BT path (f16, [n][k])

    const int bz = blockIdx.z;
    const int bb = bz / nheads, hh = bz % nheads;
    A += bb * sAb + hh * sAh;
    B += bb * sBb + hh * sBh;
    C += bb * sCb + hh * sCh;

    const int m0 = blockIdx.y * BM;
    const int n0 = blockIdx.x * BN;
    const int tid  = threadIdx.x;
    const int lane = tid & 31;
    const int wave = tid >> 5;
    const int wr = (wave >> 1) * 32;    // wave row offset in block tile
    const int wc = (wave & 1) * 32;     // wave col offset
    const int lr = lane & 15;           // lane row/col within 16
    const int lh = lane >> 4;           // lane half (0/1)

    // Prefetch one BK-step of A (and B) into LDS buffer `buf`.
    auto prefetch = [&](int buf, int k0) {
        float* Ab = As + buf * ABUF;
        #pragma unroll
        for (int u = 0; u < 4; ++u) {               // 512 float4 units, 128 thr
            int idx = tid + u * 128;
            int r = idx >> 3, c = (idx & 7) << 2;
            async_b128_to_lds(A + (m0 + r) * lda + (k0 + c), Ab + r * AST + c);
        }
        if constexpr (BT) {                         // pure copy: async DMA too
            float* Bb = Bf + buf * (BN * AST);
            #pragma unroll
            for (int u = 0; u < 4; ++u) {
                int idx = tid + u * 128;
                int r = idx >> 3, c = (idx & 7) << 2;
                async_b128_to_lds(B + (n0 + r) * ldb + (k0 + c), Bb + r * AST + c);
            }
        } else {                                    // needs transpose: via VGPRs
            _Float16* Bb = Bh + buf * (BN * BSTH);
            #pragma unroll
            for (int u = 0; u < 4; ++u) {           // 32 k-rows x 16 col-groups
                int idx = tid + u * 128;
                int r = idx >> 4, c = (idx & 15) << 2;
                v4f fv = *(const v4f*)(B + (k0 + r) * ldb + (n0 + c));
                Bb[(c + 0) * BSTH + r] = (_Float16)fv.x;
                Bb[(c + 1) * BSTH + r] = (_Float16)fv.y;
                Bb[(c + 2) * BSTH + r] = (_Float16)fv.z;
                Bb[(c + 3) * BSTH + r] = (_Float16)fv.w;
            }
        }
    };

    v8f acc[2][2] = {};
    const int nk = K / BK;

    prefetch(0, 0);
    wait_asynccnt0();
    __syncthreads();

    for (int t = 0; t < nk; ++t) {
        const int p = t & 1;
        if (t + 1 < nk) prefetch(1 - p, (t + 1) * BK);   // overlap with compute

        // ---- fragments in contiguous k order (k = 16*lh + s), same for A and B ----
        // A 16x32: lane(l&15)=row M; halves 0..15 = K[lh*16..+15]
        // B 32x16: lane(l&15)=col N; halves 0..15 = K[lh*16..+15]
        const float* Ab = As + p * ABUF;
        v16h af[2], bfr[2];
        #pragma unroll
        for (int i = 0; i < 2; ++i) {
            const float* ap = Ab + (wr + i * 16 + lr) * AST;
            v4f f0 = *(const v4f*)(ap + lh * 16);
            v4f f1 = *(const v4f*)(ap + lh * 16 + 4);
            v4f f2 = *(const v4f*)(ap + lh * 16 + 8);
            v4f f3 = *(const v4f*)(ap + lh * 16 + 12);
            v8h lo = __builtin_shufflevector(__builtin_convertvector(f0, v4h),
                                             __builtin_convertvector(f1, v4h),
                                             0,1,2,3,4,5,6,7);
            v8h hi = __builtin_shufflevector(__builtin_convertvector(f2, v4h),
                                             __builtin_convertvector(f3, v4h),
                                             0,1,2,3,4,5,6,7);
            af[i] = __builtin_shufflevector(lo, hi, 0,1,2,3,4,5,6,7,8,9,10,11,12,13,14,15);
        }
        if constexpr (BT) {
            const float* Bb = Bf + p * (BN * AST);
            #pragma unroll
            for (int i = 0; i < 2; ++i) {
                const float* bp = Bb + (wc + i * 16 + lr) * AST + lh * 16;
                v4f f0 = *(const v4f*)(bp + 0);
                v4f f1 = *(const v4f*)(bp + 4);
                v4f f2 = *(const v4f*)(bp + 8);
                v4f f3 = *(const v4f*)(bp + 12);
                v8h lo = __builtin_shufflevector(__builtin_convertvector(f0, v4h),
                                                 __builtin_convertvector(f1, v4h),
                                                 0,1,2,3,4,5,6,7);
                v8h hi = __builtin_shufflevector(__builtin_convertvector(f2, v4h),
                                                 __builtin_convertvector(f3, v4h),
                                                 0,1,2,3,4,5,6,7);
                bfr[i] = __builtin_shufflevector(lo, hi, 0,1,2,3,4,5,6,7,8,9,10,11,12,13,14,15);
            }
        } else {
            const _Float16* Bb = Bh + p * (BN * BSTH);
            #pragma unroll
            for (int i = 0; i < 2; ++i) {
                const _Float16* bp = Bb + (wc + i * 16 + lr) * BSTH + lh * 16;
                v8h b0 = *(const v8h*)(bp);
                v8h b1 = *(const v8h*)(bp + 8);
                bfr[i] = __builtin_shufflevector(b0, b1, 0,1,2,3,4,5,6,7,8,9,10,11,12,13,14,15);
            }
        }

        #pragma unroll
        for (int i = 0; i < 2; ++i)
            #pragma unroll
            for (int j = 0; j < 2; ++j)
                acc[i][j] = __builtin_amdgcn_wmma_f32_16x16x32_f16(
                    false, af[i], false, bfr[j], (short)0, acc[i][j], false, false);

        wait_asynccnt0();    // prefetch of t+1 landed
        __syncthreads();     // (also fences the ds_store transpose path)
    }

    // ---- epilogue: C/D layout: VGPR r -> M = r + 8*lh ; lane(l&15) -> N ----
    // 32-bit index math: every m*ldc+n here is < 2^31.
    #pragma unroll
    for (int i = 0; i < 2; ++i) {
        #pragma unroll
        for (int j = 0; j < 2; ++j) {
            const int n = n0 + wc + j * 16 + lr;
            float bv = 0.f;
            if constexpr (EPI >= 1) bv = bias[n];
            #pragma unroll
            for (int r = 0; r < 8; ++r) {
                const int m = m0 + wr + i * 16 + lh * 8 + r;
                float val = acc[i][j][r];
                if constexpr (EPI >= 1) val += bv;
                if constexpr (EPI == 2) val = fmaxf(val, 0.f);
                C[m * ldc + n] = val;
            }
        }
    }
}

// ---------------------------------------------------------------------------
// Embedding + sinusoidal positional encoding: x[row] = emb[tok[row]] + pe[s]
// ---------------------------------------------------------------------------
__global__ __launch_bounds__(128)
void embed_pe_kernel(const int* __restrict__ tok, const float* __restrict__ emb,
                     float* __restrict__ x)
{
    const int row = blockIdx.x;
    const int s = row & (SEQ - 1);
    const int t = tok[row];
    const float* ep = emb + (long)t * DMODEL;
    float* xp = x + (long)row * DMODEL;
    for (int d = threadIdx.x; d < DMODEL; d += 128) {
        float freq = __expf((float)(d & ~1) * (-9.210340371976184f / (float)DMODEL));
        float ang  = (float)s * freq;
        float pe   = (d & 1) ? __cosf(ang) : __sinf(ang);
        xp[d] = ep[d] + pe;
    }
}

// ---------------------------------------------------------------------------
// In-place masked softmax over scores [B,H,S,S]; scale = 1/sqrt(64)
// ---------------------------------------------------------------------------
__global__ __launch_bounds__(128)
void softmax_kernel(float* __restrict__ sc, const int* __restrict__ ktok, int causal)
{
    const int row = blockIdx.x, tid = threadIdx.x;
    const int qi = row & (SEQ - 1);
    const int b  = row / (NHEAD * SEQ);
    float* p = sc + (long)row * SEQ;
    const int* kt = ktok + (long)b * SEQ;

    float v[4];
    float mx = -3.4e38f;
    #pragma unroll
    for (int i = 0; i < 4; ++i) {
        int j = tid + 128 * i;
        float x = p[j] * 0.125f;          // 1/sqrt(hd)
        if (kt[j] == 0) x -= 1e10f;       // key padding mask
        if (causal && j > qi) x -= 1e10f; // causal mask
        v[i] = x;
        mx = fmaxf(mx, x);
    }
    __shared__ float red[128];
    red[tid] = mx; __syncthreads();
    for (int st = 64; st > 0; st >>= 1) { if (tid < st) red[tid] = fmaxf(red[tid], red[tid + st]); __syncthreads(); }
    mx = red[0]; __syncthreads();

    float s = 0.f;
    #pragma unroll
    for (int i = 0; i < 4; ++i) { v[i] = __expf(v[i] - mx); s += v[i]; }
    red[tid] = s; __syncthreads();
    for (int st = 64; st > 0; st >>= 1) { if (tid < st) red[tid] += red[tid + st]; __syncthreads(); }
    const float inv = 1.f / red[0];
    #pragma unroll
    for (int i = 0; i < 4; ++i) { int j = tid + 128 * i; p[j] = v[i] * inv; }
}

// ---------------------------------------------------------------------------
// out = LayerNorm(x + y), row length DMODEL
// ---------------------------------------------------------------------------
__global__ __launch_bounds__(128)
void add_ln_kernel(const float* __restrict__ x, const float* __restrict__ y,
                   float* __restrict__ o)
{
    const int row = blockIdx.x, tid = threadIdx.x;
    const float* xp = x + (long)row * DMODEL;
    const float* yp = y + (long)row * DMODEL;
    float* op = o + (long)row * DMODEL;

    float v[4]; float s = 0.f;
    #pragma unroll
    for (int i = 0; i < 4; ++i) { int j = tid + 128 * i; v[i] = xp[j] + yp[j]; s += v[i]; }
    __shared__ float red[128];
    red[tid] = s; __syncthreads();
    for (int st = 64; st > 0; st >>= 1) { if (tid < st) red[tid] += red[tid + st]; __syncthreads(); }
    const float mean = red[0] * (1.f / DMODEL);
    __syncthreads();
    float s2 = 0.f;
    #pragma unroll
    for (int i = 0; i < 4; ++i) { float d = v[i] - mean; s2 += d * d; }
    red[tid] = s2; __syncthreads();
    for (int st = 64; st > 0; st >>= 1) { if (tid < st) red[tid] += red[tid + st]; __syncthreads(); }
    const float inv = rsqrtf(red[0] * (1.f / DMODEL) + 1e-5f);
    #pragma unroll
    for (int i = 0; i < 4; ++i) { int j = tid + 128 * i; op[j] = (v[i] - mean) * inv; }
}

// ---------------------------------------------------------------------------
// Host-side helpers
// ---------------------------------------------------------------------------
static void gemm(hipStream_t st, const float* A, const float* B, const float* bias,
                 float* C, int M, int N, int K, int lda, int ldb, int ldc,
                 bool bt, int epi,
                 int zb = 1, int nh = 1,
                 long sAb = 0, long sAh = 0, long sBb = 0, long sBh = 0,
                 long sCb = 0, long sCh = 0)
{
    dim3 g((unsigned)(N / 64), (unsigned)(M / 64), (unsigned)zb), t(128);
    if (bt)
        gemm_wmma_kernel<true, 0><<<g, t, 0, st>>>(A, B, bias, C, K, lda, ldb, ldc, nh, sAb, sAh, sBb, sBh, sCb, sCh);
    else if (epi == 0)
        gemm_wmma_kernel<false, 0><<<g, t, 0, st>>>(A, B, bias, C, K, lda, ldb, ldc, nh, sAb, sAh, sBb, sBh, sCb, sCh);
    else if (epi == 1)
        gemm_wmma_kernel<false, 1><<<g, t, 0, st>>>(A, B, bias, C, K, lda, ldb, ldc, nh, sAb, sAh, sBb, sBh, sCb, sCh);
    else
        gemm_wmma_kernel<false, 2><<<g, t, 0, st>>>(A, B, bias, C, K, lda, ldb, ldc, nh, sAb, sAh, sBb, sBh, sCb, sCh);
}

// Full multi-head attention: aout = MHA(xq, xkv) @ Wo
static void attention(hipStream_t st,
                      const float* xq, const float* xkv,
                      const float* Wq, const float* Wk, const float* Wv, const float* Wo,
                      const int* keytok, int causal,
                      float* q, float* k, float* v, float* scores, float* attn, float* aout)
{
    const long SD = (long)SEQ * DMODEL;   // per-batch stride of q/k/v
    const long SS = (long)SEQ * SEQ;      // per-head stride of scores

    gemm(st, xq,  Wq, nullptr, q, ROWS, DMODEL, DMODEL, DMODEL, DMODEL, DMODEL, false, 0);
    gemm(st, xkv, Wk, nullptr, k, ROWS, DMODEL, DMODEL, DMODEL, DMODEL, DMODEL, false, 0);
    gemm(st, xkv, Wv, nullptr, v, ROWS, DMODEL, DMODEL, DMODEL, DMODEL, DMODEL, false, 0);

    // scores[b,h] = q[b,:,h*64:..] @ k[b,:,h*64:..]^T   (M=S, N=S, K=64)
    gemm(st, q, k, nullptr, scores, SEQ, SEQ, HDIM, DMODEL, DMODEL, SEQ, true, 0,
         BATCH * NHEAD, NHEAD, SD, HDIM, SD, HDIM, (long)NHEAD * SS, SS);

    softmax_kernel<<<BATCH * NHEAD * SEQ, 128, 0, st>>>(scores, keytok, causal);

    // attn[b,:,h*64:..] = probs[b,h] @ v[b,:,h*64:..]   (M=S, N=64, K=S)
    gemm(st, scores, v, nullptr, attn, SEQ, HDIM, SEQ, SEQ, DMODEL, DMODEL, false, 0,
         BATCH * NHEAD, NHEAD, (long)NHEAD * SS, SS, SD, HDIM, SD, HDIM);

    gemm(st, attn, Wo, nullptr, aout, ROWS, DMODEL, DMODEL, DMODEL, DMODEL, DMODEL, false, 0);
}

extern "C" void kernel_launch(void* const* d_in, const int* in_sizes, int n_in,
                              void* d_out, int out_size, void* d_ws, size_t ws_size,
                              hipStream_t stream)
{
    (void)in_sizes; (void)n_in; (void)out_size; (void)ws_size;

    const int*   src     = (const int*)d_in[0];
    const int*   tgt     = (const int*)d_in[1];
    const float* emb     = (const float*)d_in[2];
    const float* enc_Wq  = (const float*)d_in[3];
    const float* enc_Wk  = (const float*)d_in[4];
    const float* enc_Wv  = (const float*)d_in[5];
    const float* enc_Wo  = (const float*)d_in[6];
    const float* dec_Wq1 = (const float*)d_in[7];
    const float* dec_Wk1 = (const float*)d_in[8];
    const float* dec_Wv1 = (const float*)d_in[9];
    const float* dec_Wo1 = (const float*)d_in[10];
    const float* dec_Wq2 = (const float*)d_in[11];
    const float* dec_Wk2 = (const float*)d_in[12];
    const float* dec_Wv2 = (const float*)d_in[13];
    const float* dec_Wo2 = (const float*)d_in[14];
    const float* enc_W1  = (const float*)d_in[15];
    const float* enc_b1  = (const float*)d_in[16];
    const float* enc_W2  = (const float*)d_in[17];
    const float* enc_b2  = (const float*)d_in[18];
    const float* dec_W1  = (const float*)d_in[19];
    const float* dec_b1  = (const float*)d_in[20];
    const float* dec_W2  = (const float*)d_in[21];
    const float* dec_b2  = (const float*)d_in[22];
    const float* Wout    = (const float*)d_in[23];
    const float* bout    = (const float*)d_in[24];
    float* out = (float*)d_out;

    // Workspace bump allocation (~88 MB of f32)
    float* p = (float*)d_ws;
    const long MD = (long)ROWS * DMODEL;            // 2048*512
    float* xe     = p; p += MD;                     // encoder activations / output
    float* y0     = p; p += MD;                     // decoder embedded input
    float* q      = p; p += MD;
    float* k      = p; p += MD;
    float* v      = p; p += MD;
    float* attn   = p; p += MD;
    float* a      = p; p += MD;                     // post-Wo attention output
    float* t1     = p; p += MD;
    float* t2     = p; p += MD;
    float* hbuf   = p; p += (long)ROWS * DFFN;      // FFN hidden
    float* scores = p; p += (long)BATCH * NHEAD * SEQ * SEQ;

    const long DD  = (long)DMODEL * DMODEL;
    const long DF  = (long)DMODEL * DFFN;

    // Embedding + positional encoding
    embed_pe_kernel<<<ROWS, 128, 0, stream>>>(src, emb, xe);
    embed_pe_kernel<<<ROWS, 128, 0, stream>>>(tgt, emb, y0);

    // ---------------- Encoder (layers chain through xe) ----------------
    for (int i = 0; i < NLAYER; ++i) {
        attention(stream, xe, xe,
                  enc_Wq + i * DD, enc_Wk + i * DD, enc_Wv + i * DD, enc_Wo + i * DD,
                  src, /*causal=*/0, q, k, v, scores, attn, a);
        add_ln_kernel<<<ROWS, 128, 0, stream>>>(xe, a, t1);                      // out = LN(x + a)
        gemm(stream, t1, enc_W1 + i * DF, enc_b1 + (long)i * DFFN, hbuf,
             ROWS, DFFN, DMODEL, DMODEL, DFFN, DFFN, false, 2);                  // relu(out@W1+b1)
        gemm(stream, hbuf, enc_W2 + i * DF, enc_b2 + (long)i * DMODEL, t2,
             ROWS, DMODEL, DFFN, DFFN, DMODEL, DMODEL, false, 1);                // f = h@W2+b2
        add_ln_kernel<<<ROWS, 128, 0, stream>>>(a, t2, xe);                      // x = LN(a + f)  (ref quirk)
    }

    // ---------------- Decoder ----------------
    // Reference quirk: every decoder layer reads the ORIGINAL y0 and overwrites
    // y; y never feeds forward. Hence only layer L-1 affects the logits, so we
    // compute exactly that layer (mathematically identical, 6x less work).
    {
        const int i = NLAYER - 1;
        attention(stream, y0, y0,
                  dec_Wq1 + i * DD, dec_Wk1 + i * DD, dec_Wv1 + i * DD, dec_Wo1 + i * DD,
                  tgt, /*causal=*/1, q, k, v, scores, attn, a);
        add_ln_kernel<<<ROWS, 128, 0, stream>>>(y0, a, t1);                      // o1 = LN(y0 + a1)
        attention(stream, t1, xe,
                  dec_Wq2 + i * DD, dec_Wk2 + i * DD, dec_Wv2 + i * DD, dec_Wo2 + i * DD,
                  src, /*causal=*/0, q, k, v, scores, attn, a);
        add_ln_kernel<<<ROWS, 128, 0, stream>>>(t1, a, t2);                      // o2 = LN(o1 + a2)
        gemm(stream, t2, dec_W1 + i * DF, dec_b1 + (long)i * DFFN, hbuf,
             ROWS, DFFN, DMODEL, DMODEL, DFFN, DFFN, false, 2);
        gemm(stream, hbuf, dec_W2 + i * DF, dec_b2 + (long)i * DMODEL, a,
             ROWS, DMODEL, DFFN, DFFN, DMODEL, DMODEL, false, 1);                // f
        add_ln_kernel<<<ROWS, 128, 0, stream>>>(t2, a, t1);                      // y = LN(o2 + f)
    }

    // Output projection: logits = y @ Wout + bout   (2048 x 32000, K=512)
    gemm(stream, t1, Wout, bout, out, ROWS, VOCAB, DMODEL, DMODEL, VOCAB, VOCAB, false, 1);
}
